// QKVCrossAttentionLegacy_12094627906381
// MI455X (gfx1250) — matrix operand order repaired and checked by reading, not compiled, not run on hardware
//
#include <hip/hip_runtime.h>
#include <math.h>

#define HEADS 32
#define CH    64
#define SEQ   2048
#define WAVES 4
#define QT    2                       // query tiles per wave
#define TQ_WG (WAVES * QT * 16)      // 128 queries per workgroup
#define SKV   64                     // kv-positions processed per iteration

typedef __attribute__((ext_vector_type(16))) __bf16 v16bf;
typedef __attribute__((ext_vector_type(8)))  float  v8f;

union V16U { unsigned int u[8]; v16bf v; };

#if __has_builtin(__builtin_amdgcn_exp2f)
#define EXP2(x) __builtin_amdgcn_exp2f(x)   // raw v_exp_f32: no denorm-range fixup
#else
#define EXP2(x) exp2f(x)
#endif

__device__ __forceinline__ unsigned int rne_bias(unsigned int u) {
  // round-to-nearest-even bias for fp32 -> bf16 truncation
  return u + 0x7FFFu + ((u >> 16) & 1u);
}
__device__ __forceinline__ unsigned short f2bf(float f) {
  return (unsigned short)(rne_bias(__float_as_uint(f)) >> 16);
}
__device__ __forceinline__ unsigned int packbf(float a, float b) {
#if __has_builtin(__builtin_amdgcn_perm)
  // v_perm_b32: bytes {b.hi16, a.hi16} in one instruction
  return __builtin_amdgcn_perm(rne_bias(__float_as_uint(b)),
                               rne_bias(__float_as_uint(a)), 0x07060302u);
#else
  return (unsigned int)f2bf(a) | ((unsigned int)f2bf(b) << 16);
#endif
}
// 16-bit A-matrix layout (contiguous k = 16*hl + s): K-pair base for VGPR j, lane-half hl
__device__ __forceinline__ int akb(int j, int hl) {
  return hl * 16 + 2 * j;
}

__global__ __launch_bounds__(128)
void flash_attn_bf16_wmma(const float* __restrict__ q,
                          const float* __restrict__ kv,
                          float* __restrict__ out) {
  __shared__ unsigned short Ksh[2][SKV * CH];          // K tile, transposed [s][c], dbl-buffered
  __shared__ unsigned short Vsh[2][CH * SKV];          // V tile, natural   [c][s], dbl-buffered
  __shared__ unsigned short Psh[WAVES][QT][16 * SKV];  // per-(wave,qtile) P' tile [t][s]

  const int tid  = threadIdx.x;
  const int lane = tid & 31;
  const int wv   = tid >> 5;
  const int ln   = lane & 15;   // N index (t) / M-within-half
  const int hl   = lane >> 4;   // lane half

  const int h  = blockIdx.x >> 4;                       // 16 query-blocks per head
  const int t0 = (blockIdx.x & 15) * TQ_WG + wv * (QT * 16);

  // 1/sqrt(ch) (both q&k scales folded) * log2(e) so softmax uses exp2
  const float QSCALE = 0.125f * 1.44269504088896340736f;

  const float* qh = q  + (size_t)h * CH * SEQ;
  const float* kh = kv + (size_t)h * (2 * CH) * SEQ;
  const float* vh = kh + (size_t)CH * SEQ;

  // ---- Q tiles as B-operands (k=c, n=t), two K=32 chunks each, loaded once ----
  V16U qb[QT][2];
#pragma unroll
  for (int qt = 0; qt < QT; ++qt) {
    const int t = t0 + qt * 16 + ln;
#pragma unroll
    for (int chunk = 0; chunk < 2; ++chunk)
#pragma unroll
      for (int j = 0; j < 8; ++j) {
        int c = chunk * 32 + hl * 16 + 2 * j;   // B layout: K-pairs adjacent in c
        float f0 = qh[(size_t)c * SEQ + t] * QSCALE;
        float f1 = qh[(size_t)(c + 1) * SEQ + t] * QSCALE;
        qb[qt][chunk].u[j] = packbf(f0, f1);
      }
  }

  // ---- cooperative KV staging: global f32 -> LDS bf16 (128 threads) ----
  const int sc  = tid >> 1;          // c row 0..63
  const int sb0 = (tid & 1) * 32;    // s sub-block
  auto stage = [&](int s0, int buf) {
    const float4* kg4 = (const float4*)(kh + (size_t)sc * SEQ + s0 + sb0);
    const float4* vg4 = (const float4*)(vh + (size_t)sc * SEQ + s0 + sb0);
#pragma unroll
    for (int i4 = 0; i4 < 8; ++i4) {
      float4 kf = kg4[i4];
      float4 vf = vg4[i4];
      int sb = sb0 + i4 * 4;
      Ksh[buf][(sb + 0) * CH + sc] = f2bf(kf.x);   // transpose into [s][c]
      Ksh[buf][(sb + 1) * CH + sc] = f2bf(kf.y);
      Ksh[buf][(sb + 2) * CH + sc] = f2bf(kf.z);
      Ksh[buf][(sb + 3) * CH + sc] = f2bf(kf.w);
      unsigned int* vrow = (unsigned int*)&Vsh[buf][sc * SKV + sb];
      vrow[0] = packbf(vf.x, vf.y);
      vrow[1] = packbf(vf.z, vf.w);
    }
  };

  float mrun[QT], lrun[QT];
  v8f o[QT][4];
#pragma unroll
  for (int qt = 0; qt < QT; ++qt) {
    mrun[qt] = -1.0e30f;
    lrun[qt] = 0.0f;
#pragma unroll
    for (int ct = 0; ct < 4; ++ct)
#pragma unroll
      for (int r = 0; r < 8; ++r) o[qt][ct][r] = 0.0f;
  }

  stage(0, 0);
  __syncthreads();
  int p = 0;

  for (int s0 = 0; s0 < SEQ; s0 += SKV) {
    // stage next tile into the other buffer while computing on this one
    if (s0 + SKV < SEQ) {
      if (s0 + 2 * SKV < SEQ) {
        __builtin_prefetch(kh + (size_t)sc * SEQ + s0 + 2 * SKV + sb0, 0, 0);
        __builtin_prefetch(vh + (size_t)sc * SEQ + s0 + 2 * SKV + sb0, 0, 0);
      }
      stage(s0 + SKV, p ^ 1);
    }
    const unsigned short* Kb = Ksh[p];
    const unsigned short* Vb = Vsh[p];

    // ---- S' = K_tile x Q : four s-tiles as M; each K A-tile feeds QT WMMAs ----
    v8f sS[QT][4];
#pragma unroll
    for (int qt = 0; qt < QT; ++qt)
#pragma unroll
      for (int st = 0; st < 4; ++st)
#pragma unroll
        for (int r = 0; r < 8; ++r) sS[qt][st][r] = 0.0f;
#pragma unroll
    for (int chunk = 0; chunk < 2; ++chunk)
#pragma unroll
      for (int st = 0; st < 4; ++st) {
        V16U ka;
#pragma unroll
        for (int j = 0; j < 8; ++j)
          ka.u[j] = *(const unsigned int*)&Kb[(st * 16 + ln) * CH + chunk * 32 + akb(j, hl)];
#pragma unroll
        for (int qt = 0; qt < QT; ++qt)
          sS[qt][st] = __builtin_amdgcn_wmma_f32_16x16x32_bf16(
              false, ka.v, false, qb[qt][chunk].v, (short)0, sS[qt][st], false, false);
      }

    // ---- online softmax per q-tile: 32 values per lane + one xor-16 exchange ----
#pragma unroll
    for (int qt = 0; qt < QT; ++qt) {
      float mx = sS[qt][0][0];
#pragma unroll
      for (int st = 0; st < 4; ++st)
#pragma unroll
        for (int r = 0; r < 8; ++r) mx = fmaxf(mx, sS[qt][st][r]);
      mx = fmaxf(mx, __shfl_xor(mx, 16, 32));
      float mn = fmaxf(mrun[qt], mx);
      float cr = EXP2(mrun[qt] - mn);
      float rs = 0.0f;
#pragma unroll
      for (int st = 0; st < 4; ++st)
#pragma unroll
        for (int r = 0; r < 8; ++r) {
          sS[qt][st][r] = EXP2(sS[qt][st][r] - mn);
          rs += sS[qt][st][r];
        }
      rs += __shfl_xor(rs, 16, 32);
      lrun[qt] = lrun[qt] * cr + rs;
      mrun[qt] = mn;
#pragma unroll
      for (int ct = 0; ct < 4; ++ct)
#pragma unroll
        for (int r = 0; r < 8; ++r) o[qt][ct][r] *= cr;

      // P': C layout (M=s,N=t) -> LDS [t][s] (same-wave, in-order DS)
      unsigned short* Pw = Psh[wv][qt];
#pragma unroll
      for (int st = 0; st < 4; ++st)
#pragma unroll
        for (int r = 0; r < 8; ++r)
          Pw[ln * SKV + st * 16 + r + 8 * hl] = f2bf(sS[qt][st][r]);
    }

    // ---- reload P' in B layout ----
    V16U pb[QT][2];
#pragma unroll
    for (int qt = 0; qt < QT; ++qt)
#pragma unroll
      for (int g = 0; g < 2; ++g)
#pragma unroll
        for (int j = 0; j < 8; ++j)
          pb[qt][g].u[j] =
              *(const unsigned int*)&Psh[wv][qt][ln * SKV + g * 32 + hl * 16 + 2 * j];

    // ---- O += V_tile x P' : each V A-tile feeds QT WMMAs ----
#pragma unroll
    for (int ct = 0; ct < 4; ++ct)
#pragma unroll
      for (int g = 0; g < 2; ++g) {
        V16U va;
#pragma unroll
        for (int j = 0; j < 8; ++j)
          va.u[j] = *(const unsigned int*)&Vb[(ct * 16 + ln) * SKV + g * 32 + akb(j, hl)];
#pragma unroll
        for (int qt = 0; qt < QT; ++qt)
          o[qt][ct] = __builtin_amdgcn_wmma_f32_16x16x32_bf16(
              false, va.v, false, pb[qt][g].v, (short)0, o[qt][ct], false, false);
      }

    __syncthreads();   // staging of next buffer done; readers of this buffer done
    p ^= 1;
  }

  // ---- epilogue: normalize, coalesced stores over t ----
  float* oh = out + (size_t)h * CH * SEQ;
#pragma unroll
  for (int qt = 0; qt < QT; ++qt) {
    const float inv = 1.0f / lrun[qt];
#pragma unroll
    for (int ct = 0; ct < 4; ++ct)
#pragma unroll
      for (int r = 0; r < 8; ++r) {
        int c = ct * 16 + r + 8 * hl;
        oh[(size_t)c * SEQ + t0 + qt * 16 + ln] = o[qt][ct][r] * inv;
      }
  }
}

extern "C" void kernel_launch(void* const* d_in, const int* in_sizes, int n_in,
                              void* d_out, int out_size, void* d_ws, size_t ws_size,
                              hipStream_t stream) {
  (void)in_sizes; (void)n_in; (void)out_size; (void)d_ws; (void)ws_size;
  const float* q  = (const float*)d_in[0];   // (32, 64, 2048) f32
  const float* kv = (const float*)d_in[1];   // (4, 1024, 2048) f32 == (32, 128, 2048)
  float* out = (float*)d_out;                // (4, 512, 2048) f32 == (32, 64, 2048)
  dim3 grid(HEADS * (SEQ / TQ_WG));          // 32 heads * 16 query blocks = 512
  dim3 block(WAVES * 32);                    // 4 waves of 32
  hipLaunchKernelGGL(flash_attn_bf16_wmma, grid, block, 0, stream, q, kv, out);
}
